// BitLinear_26912265077068
// MI455X (gfx1250) — matrix operand order repaired and checked by reading, not compiled, not run on hardware
//
#include <hip/hip_runtime.h>
#include <cstdint>

// Problem constants (match reference)
#define O_DIM 11008
#define I_DIM 4096
#define B_DIM 512
#define IW    (I_DIM / 16)   // packed int32 words per output-feature row = 256

// Tile config
#define BM 128
#define BN 128
#define BK 64
#define XS 72   // LDS row stride (halves) for x tile  (64 + 8 pad, 16B aligned)
#define WS 72   // LDS row stride (halves) for w tile

typedef __attribute__((ext_vector_type(16))) _Float16 v16h;
typedef __attribute__((ext_vector_type(8)))  float    v8f;
typedef __attribute__((ext_vector_type(4)))  int      v4i;

// Pointer types matching the async-to-LDS builtin's expected signature:
// (v4i addrspace(1)* global_src, v4i addrspace(3)* lds_dst, imm offset, imm cpol)
typedef __attribute__((address_space(1))) v4i* gptr4;
typedef __attribute__((address_space(3))) v4i* lptr4;

#if __has_builtin(__builtin_amdgcn_global_load_async_to_lds_b128)
#define HAVE_ASYNC_LDS 1
#else
#define HAVE_ASYNC_LDS 0
#endif

static __device__ __forceinline__ void wait_async0() {
#if __has_builtin(__builtin_amdgcn_s_wait_asynccnt)
  __builtin_amdgcn_s_wait_asynccnt(0);
#else
  asm volatile("s_wait_asynccnt 0x0" ::: "memory");
#endif
}

// Decode one packed int32 word (16 ternary codes, 2b each: 0->0, 1->+1, 2->-1)
// into 8 u32s, each packing two fp16 values (low half = even code).
static __device__ __forceinline__ void decode16(uint32_t w, uint32_t d[8]) {
  uint32_t mag = (w | (w >> 1)) & 0x55555555u;  // bit(2j)   = code j != 0
  uint32_t sgn = w & 0xAAAAAAAAu;               // bit(2j+1) = code j is -1
#pragma unroll
  for (int p = 0; p < 8; ++p) {
    uint32_t m = (mag >> (4 * p)) & 0x5u;  // mag bits at 0 and 2
    uint32_t s = (sgn >> (4 * p)) & 0xAu;  // sign bits at 1 and 3
    uint32_t h = ((m & 1u) ? 0x3C00u : 0u)       // lo fp16 magnitude
               | ((s & 2u) << 14)                 // lo sign -> bit 15
               | ((m & 4u) ? 0x3C000000u : 0u)    // hi fp16 magnitude
               | ((s & 8u) << 28);                // hi sign -> bit 31
    d[p] = h;
  }
}

// Assemble a v16h fragment from two 16-byte LDS chunks.
static __device__ __forceinline__ v16h load_frag(const _Float16* p0, const _Float16* p1) {
  union { uint4 u[2]; v16h h; } u;
  u.u[0] = *(const uint4*)p0;
  u.u[1] = *(const uint4*)p1;
  return u.h;
}

__global__ __launch_bounds__(256)
void bitlinear_wmma_kernel(const _Float16* __restrict__ x,
                           const uint32_t* __restrict__ pw,
                           const _Float16* __restrict__ bias,
                           _Float16* __restrict__ out) {
  const int tid  = threadIdx.x;
  const int wave = tid >> 5;
  const int lane = tid & 31;
  const int ln   = lane & 15;   // column/row-within-tile lane index
  const int g    = lane >> 4;   // lane half-group (0: lanes 0-15, 1: lanes 16-31)

  const int waveM = wave >> 2;  // 0..1, owns 64 rows of M
  const int waveN = wave & 3;   // 0..3, owns 32 cols of N

  const int n0 = blockIdx.x * BN;   // output-feature tile base
  const int m0 = blockIdx.y * BM;   // batch-row tile base

  __shared__ __align__(16) _Float16 sx[2][BM * XS];  // double-buffered x tile
  __shared__ __align__(16) _Float16 sw[BN * WS];     // decoded ternary tile

  const _Float16* xg_base = x + (size_t)m0 * I_DIM;

  // Async-stage one 128x64 fp16 x-tile into LDS buffer `buf`.
  auto stage_x = [&](int kb, int buf) {
#pragma unroll
    for (int t = 0; t < 4; ++t) {
      int c   = t * 256 + tid;
      int row = c >> 3;        // 8 x 16B chunks per row
      int c8  = c & 7;
      const _Float16* gsrc = xg_base + (size_t)row * I_DIM + kb + c8 * 8;
      _Float16*       ldst = &sx[buf][row * XS + c8 * 8];
#if HAVE_ASYNC_LDS
      __builtin_amdgcn_global_load_async_to_lds_b128(
          (gptr4)(void*)gsrc, (lptr4)(void*)ldst, /*offset=*/0, /*cpol=*/0);
#else
      *(uint4*)ldst = *(const uint4*)gsrc;
#endif
    }
  };

  v8f acc[4][2];
  {
    v8f z = {};
#pragma unroll
    for (int mt = 0; mt < 4; ++mt)
#pragma unroll
      for (int nt = 0; nt < 2; ++nt) acc[mt][nt] = z;
  }

  // Prologue: start async copy of the first x tile.
  stage_x(0, 0);

  for (int i = 0; i < I_DIM / BK; ++i) {
    const int kb  = i * BK;
    const int cur = i & 1;

    // ---- Stage + decode W tile: 128 features x 4 words, 2 words/thread ----
#pragma unroll
    for (int t = 0; t < 2; ++t) {
      int widx = t * 256 + tid;
      int f    = widx >> 2;    // feature within tile
      int ww   = widx & 3;     // word within K-step (covers 16 K values)
      uint32_t word = pw[(size_t)(n0 + f) * IW + (kb >> 4) + ww];
      uint32_t d[8];
      decode16(word, d);
      uint4 lo, hi;
      lo.x = d[0]; lo.y = d[1]; lo.z = d[2]; lo.w = d[3];
      hi.x = d[4]; hi.y = d[5]; hi.z = d[6]; hi.w = d[7];
      _Float16* dst = &sw[f * WS + ww * 16];
      *(uint4*)dst       = lo;
      *(uint4*)(dst + 8) = hi;
    }

    // Prefetch next packed-weight words (emits global_prefetch_b8).
    if (kb + BK < I_DIM) {
      __builtin_prefetch(&pw[(size_t)(n0 + (tid >> 2)) * IW + ((kb + BK) >> 4) + (tid & 3)], 0, 0);
    }

#if HAVE_ASYNC_LDS
    wait_async0();           // x tile `cur` fully landed in LDS
#endif
    __syncthreads();         // + all waves' decoded weights visible

    // Kick off the next x tile into the other buffer while we compute.
    if (kb + BK < I_DIM) stage_x(kb + BK, cur ^ 1);

    // ---- Compute: two 32-deep K chunks, 8 WMMAs each ----
    const _Float16* sxc = sx[cur];
#pragma unroll
    for (int kk = 0; kk < BK; kk += 32) {
      v16h a[4], b[2];
#pragma unroll
      for (int mt = 0; mt < 4; ++mt) {
        // A 16x32 f16 layout: lanes 0-15 hold K {0..7,16..23}, lanes 16-31 hold {8..15,24..31}
        int row  = waveM * 64 + mt * 16 + ln;
        int base = row * XS + kk + g * 8;
        a[mt] = load_frag(&sxc[base], &sxc[base + 16]);
      }
#pragma unroll
      for (int nt = 0; nt < 2; ++nt) {
        // B 32x16 f16 layout: lanes 0-15 hold K {0..7,16..23}, lanes 16-31 hold {8..15,24..31}
        int col  = waveN * 32 + nt * 16 + ln;
        int base = col * WS + kk + g * 8;
        b[nt] = load_frag(&sw[base], &sw[base + 16]);
      }
#pragma unroll
      for (int mt = 0; mt < 4; ++mt)
#pragma unroll
        for (int nt = 0; nt < 2; ++nt)
          acc[mt][nt] = __builtin_amdgcn_wmma_f32_16x16x32_f16(
              /*neg_a=*/false, a[mt], /*neg_b=*/false, b[nt],
              /*c_mod=*/(short)0, acc[mt][nt],
              /*reuse_a=*/false, /*reuse_b=*/false);
    }

    __syncthreads();  // protect sw (and retire readers) before next decode
  }

  // ---- Epilogue: add bias, cast to fp16, store ----
  // C/D layout: lane group g, column ln; VGPR r holds M = r + 8*g.
#pragma unroll
  for (int nt = 0; nt < 2; ++nt) {
    int col = n0 + waveN * 32 + nt * 16 + ln;
    float bf = (float)bias[col];
#pragma unroll
    for (int mt = 0; mt < 4; ++mt) {
      int rowb = m0 + waveM * 64 + mt * 16 + g * 8;
#pragma unroll
      for (int r = 0; r < 8; ++r) {
        out[(size_t)(rowb + r) * O_DIM + col] = (_Float16)(acc[mt][nt][r] + bf);
      }
    }
  }
}

extern "C" void kernel_launch(void* const* d_in, const int* in_sizes, int n_in,
                              void* d_out, int out_size, void* d_ws, size_t ws_size,
                              hipStream_t stream) {
  (void)in_sizes; (void)n_in; (void)out_size; (void)d_ws; (void)ws_size;
  const _Float16*  x    = (const _Float16*)d_in[0];
  const uint32_t*  pw   = (const uint32_t*)d_in[1];
  const _Float16*  bias = (const _Float16*)d_in[2];
  _Float16*        out  = (_Float16*)d_out;

  dim3 grid(O_DIM / BN, B_DIM / BM);  // (86, 4)
  bitlinear_wmma_kernel<<<grid, 256, 0, stream>>>(x, pw, bias, out);
}
